// ldbsdeRNN_57269093925271
// MI455X (gfx1250) — matrix operand order repaired and checked by reading, not compiled, not run on hardware
//
#include <hip/hip_runtime.h>

#define SIGMA   0.2f
#define DD      100
#define ETA     512
#define NSTEPS  50
#define NT1     (NSTEPS + 1)   // 51
#define BATCH   4096
#define BT      64             // batch rows per workgroup
#define TPB     256            // 8 waves (wave32)

typedef __attribute__((ext_vector_type(16))) __bf16 v16bf;
typedef __attribute__((ext_vector_type(8)))  float  v8f;

// Packed weight sizes (uints)
#define WHB_N   (32 * 512 * 8)   // W_h:    K=512 (32 kh), N=512
#define WXB_N   (8 * 512 * 8)    // W_X_x:  K=128 (8 kh),  N=512
#define WXTB_N  (32 * 128 * 8)   // W_X_x^T:K=512 (32 kh), N=128

__device__ __forceinline__ unsigned short f2bf(float f) {
  unsigned int u = __float_as_uint(f);
  u += 0x7FFFu + ((u >> 16) & 1u);          // round-to-nearest-even
  return (unsigned short)(u >> 16);
}
__device__ __forceinline__ float bfLo(unsigned int u) {
  return __uint_as_float(u << 16);
}
__device__ __forceinline__ float bfHi(unsigned int u) {
  return __uint_as_float(u & 0xFFFF0000u);
}
__device__ __forceinline__ unsigned int pack2bf(float lo, float hi) {
  return (unsigned int)f2bf(lo) | ((unsigned int)f2bf(hi) << 16);
}

// gfx1250 hardware tanh (v_tanh_f32) with safe fallback
__device__ __forceinline__ float htanh(float x) {
#if __has_builtin(__builtin_amdgcn_tanhf)
  return __builtin_amdgcn_tanhf(x);
#elif __has_builtin(__builtin_amdgcn_tanh_f32)
  return __builtin_amdgcn_tanh_f32(x);
#else
  return tanhf(x);
#endif
}

// Async global->LDS copy (CDNA5), tracked by ASYNCcnt. EXEC-masked per lane.
__device__ __forceinline__ void asyncLoadF32(const float* g, unsigned lds_off) {
  asm volatile("global_load_async_to_lds_b32 %0, %1, off"
               :: "v"(lds_off), "v"(g) : "memory");
}
__device__ __forceinline__ void waitAsync0() {
#if __has_builtin(__builtin_amdgcn_s_wait_asynccnt)
  __builtin_amdgcn_s_wait_asynccnt(0);
#else
  asm volatile("s_wait_asynccnt 0x0" ::: "memory");
#endif
}

union Frag {
  uint4 q[2];
  v16bf v;
};

// A-fragment 16x32 bf16 from LDS (row-major, ldk elements/row).
__device__ __forceinline__ v16bf loadA(const unsigned short* Sm, int row0,
                                       int k0, int ldk, int lane) {
  const int half = lane >> 4;
  const unsigned short* p = Sm + (row0 + (lane & 15)) * ldk + k0 + half * 16;
  Frag f;
  f.q[0] = *(const uint4*)(p);
  f.q[1] = *(const uint4*)(p + 8);
  return f.v;
}

// B-fragment 32x16 bf16 from packed global weights laid out [kh][n][v] (uint).
__device__ __forceinline__ v16bf loadB(const unsigned int* __restrict__ Wp,
                                       int kh, int ncol, int ncols) {
  const unsigned int* p = Wp + (((size_t)kh * ncols + ncol) << 3);
  Frag f;
  f.q[0] = *(const uint4*)(p);
  f.q[1] = *(const uint4*)(p + 4);
  return f.v;
}

__device__ __forceinline__ v8f wmma_bf16(v16bf a, v16bf b, v8f c) {
  return __builtin_amdgcn_wmma_f32_16x16x32_bf16(false, a, false, b,
                                                 (short)0, c, false, false);
}

// ---------------- weight repack (runs once per launch) ----------------
__global__ __launch_bounds__(TPB) void prep_kernel(
    const float* __restrict__ W_X, const float* __restrict__ W_h,
    unsigned int* __restrict__ WhB, unsigned int* __restrict__ WXB,
    unsigned int* __restrict__ WXTB) {
  const int i = blockIdx.x * TPB + threadIdx.x;
  if (i < WHB_N) {
    const int v = i & 7, n = (i >> 3) & 511, kh = i >> 12;
    const int k = kh * 16 + 2 * v;
    WhB[i] = pack2bf(W_h[k * ETA + n], W_h[(k + 1) * ETA + n]);
  } else if (i < WHB_N + WXB_N) {
    const int j = i - WHB_N;
    const int v = j & 7, n = (j >> 3) & 511, kh = j >> 12;
    const int k = kh * 16 + 2 * v;                 // 0..126 (padded to 128)
    const float lo = (k     < DD) ? W_X[(1 + k) * ETA + n] : 0.f;
    const float hi = (k + 1 < DD) ? W_X[(2 + k) * ETA + n] : 0.f;
    WXB[j] = pack2bf(lo, hi);
  } else {
    const int j = i - (WHB_N + WXB_N);
    const int v = j & 7, n = (j >> 3) & 127, kh = j >> 10;
    const int k = kh * 16 + 2 * v;                 // 0..510
    const float lo = (n < DD) ? W_X[(1 + n) * ETA + k]     : 0.f;
    const float hi = (n < DD) ? W_X[(1 + n) * ETA + k + 1] : 0.f;
    WXTB[j] = pack2bf(lo, hi);
  }
}

// ---------------- main scan kernel ----------------
__global__ __launch_bounds__(TPB, 1) void rnn_kernel(
    const float* __restrict__ tArr, const float* __restrict__ X,
    const float* __restrict__ h_m1, const float* __restrict__ W_X,
    const float* __restrict__ b_h, const float* __restrict__ W_y,
    const float* __restrict__ b_y,
    const unsigned int* __restrict__ WhB, const unsigned int* __restrict__ WXB,
    const unsigned int* __restrict__ WXTB,
    float* __restrict__ Yout, float* __restrict__ Zout) {
  __shared__ __attribute__((aligned(16))) unsigned short hS[BT * ETA]; // 64 KB
  __shared__ __attribute__((aligned(16))) unsigned short gS[BT * ETA]; // 64 KB
  __shared__ __attribute__((aligned(16))) unsigned short xS[BT * 128]; // 16 KB
  __shared__ __attribute__((aligned(16))) float xF[BT * 128];          // 32 KB (async staging)
  __shared__ float bC[ETA], wtC[ETA], wyC[ETA];
  __shared__ float tC[NT1];

  const int tid   = threadIdx.x;
  const int lane  = tid & 31;
  const int wave  = tid >> 5;
  const int half  = lane >> 4;
  const int ncl   = lane & 15;
  const int bbase = blockIdx.x * BT;

  // Per-thread X staging geometry: 4 threads per batch row, 32 d each.
  const int xrow = tid >> 2;
  const int xd0  = (tid & 3) * 32;
  const float* xgbase = X + (size_t)(bbase + xrow) * (DD * NT1);

  // ---- kick off async prefetch of X[:, :, 0] while we init LDS ----
  {
#pragma unroll 4
    for (int j = 0; j < 32; ++j) {
      const int d = xd0 + j;
      if (d < DD)
        asyncLoadF32(xgbase + (size_t)d * NT1,
                     (unsigned)(size_t)&xF[xrow * 128 + d]);
    }
  }

  for (int k = tid; k < ETA; k += TPB) {
    bC[k]  = b_h[k];
    wtC[k] = W_X[k];   // W_X_t = W_X[0, :]
    wyC[k] = W_y[k];   // w_y  = W_y[:, 0]
  }
  if (tid < NT1) tC[tid] = tArr[tid];
  for (int i = tid; i < BT * ETA; i += TPB)     // h0 = broadcast(h_m1)
    hS[i] = f2bf(h_m1[i & (ETA - 1)]);
  waitAsync0();
  __syncthreads();

  const float by = b_y[0];
  const v8f vzero = {0.f, 0.f, 0.f, 0.f, 0.f, 0.f, 0.f, 0.f};

  for (int s = 0; s < NT1; ++s) {
    // ---- convert staged Xn (f32, in xF) -> bf16 xS, D padded to 128 ----
    {
      const float4* fp = (const float4*)(xF + xrow * 128 + xd0);
      uint4* op = (uint4*)(xS + xrow * 128 + xd0);
#pragma unroll
      for (int g4 = 0; g4 < 4; ++g4) {
        const float4 f0 = fp[g4 * 2];
        const float4 f1 = fp[g4 * 2 + 1];
        const int db = xd0 + g4 * 8;
        const float a0 = (db + 0 < DD) ? f0.x : 0.f;
        const float a1 = (db + 1 < DD) ? f0.y : 0.f;
        const float a2 = (db + 2 < DD) ? f0.z : 0.f;
        const float a3 = (db + 3 < DD) ? f0.w : 0.f;
        const float a4 = (db + 4 < DD) ? f1.x : 0.f;
        const float a5 = (db + 5 < DD) ? f1.y : 0.f;
        const float a6 = (db + 6 < DD) ? f1.z : 0.f;
        const float a7 = (db + 7 < DD) ? f1.w : 0.f;
        uint4 o;
        o.x = pack2bf(a0, a1);
        o.y = pack2bf(a2, a3);
        o.z = pack2bf(a4, a5);
        o.w = pack2bf(a6, a7);
        op[g4] = o;
      }
    }
    __syncthreads();   // xS ready; xF free for next prefetch

    // ---- async prefetch next step's Xn into xF (hidden under the GEMM) ----
    if (s + 1 < NT1) {
#pragma unroll 4
      for (int j = 0; j < 32; ++j) {
        const int d = xd0 + j;
        if (d < DD)
          asyncLoadF32(xgbase + (size_t)d * NT1 + (s + 1),
                       (unsigned)(size_t)&xF[xrow * 128 + d]);
      }
    }

    const float tn = tC[s];

    // ---- pre = h@W_h + Xn@W_X_x  (wave covers N cols [wave*64, wave*64+64)) ----
    v8f acc[4][4];
#pragma unroll
    for (int mt = 0; mt < 4; ++mt)
#pragma unroll
      for (int nt = 0; nt < 4; ++nt) acc[mt][nt] = vzero;

#pragma unroll 2
    for (int kc = 0; kc < 16; ++kc) {           // K = 512
      v16bf a[4], bb[4];
#pragma unroll
      for (int mt = 0; mt < 4; ++mt)
        a[mt] = loadA(hS, mt * 16, kc * 32, ETA, lane);
#pragma unroll
      for (int nt = 0; nt < 4; ++nt)
        bb[nt] = loadB(WhB, kc * 2 + half, wave * 64 + nt * 16 + ncl, ETA);
#pragma unroll
      for (int mt = 0; mt < 4; ++mt)
#pragma unroll
        for (int nt = 0; nt < 4; ++nt)
          acc[mt][nt] = wmma_bf16(a[mt], bb[nt], acc[mt][nt]);
    }
#pragma unroll
    for (int kc = 0; kc < 4; ++kc) {            // K = 128 (X part)
      v16bf a[4], bb[4];
#pragma unroll
      for (int mt = 0; mt < 4; ++mt)
        a[mt] = loadA(xS, mt * 16, kc * 32, 128, lane);
#pragma unroll
      for (int nt = 0; nt < 4; ++nt)
        bb[nt] = loadB(WXB, kc * 2 + half, wave * 64 + nt * 16 + ncl, ETA);
#pragma unroll
      for (int mt = 0; mt < 4; ++mt)
#pragma unroll
        for (int nt = 0; nt < 4; ++nt)
          acc[mt][nt] = wmma_bf16(a[mt], bb[nt], acc[mt][nt]);
    }
    __syncthreads();   // all waves done reading hS before overwrite

    // ---- bias + hw tanh; write h_new AND g = (1-h^2)*w_y back (bf16) ----
#pragma unroll
    for (int nt = 0; nt < 4; ++nt) {
      const int n = wave * 64 + nt * 16 + ncl;
      const float bias = bC[n] + tn * wtC[n];
      const float wyn = wyC[n];
#pragma unroll
      for (int mt = 0; mt < 4; ++mt)
#pragma unroll
        for (int r = 0; r < 8; ++r) {
          const int row = mt * 16 + half * 8 + r;  // C layout: VGPR r -> M=r / 8+r
          const float hv = htanh(acc[mt][nt][r] + bias);
          hS[row * ETA + n] = f2bf(hv);
          gS[row * ETA + n] = f2bf((1.f - hv * hv) * wyn);
        }
    }
    __syncthreads();

    // ---- y = h@W_y + b_y : vectorized read + 4-wide shuffle reduce ----
    {
      const int row = tid >> 2, seg = tid & 3;
      const uint4* hp = (const uint4*)(hS + row * ETA + seg * 128);
      const float* wp = wyC + seg * 128;
      float sum = 0.f;
#pragma unroll 4
      for (int q = 0; q < 16; ++q) {
        const uint4 u = hp[q];
        const float* w = wp + q * 8;
        sum += bfLo(u.x) * w[0] + bfHi(u.x) * w[1];
        sum += bfLo(u.y) * w[2] + bfHi(u.y) * w[3];
        sum += bfLo(u.z) * w[4] + bfHi(u.z) * w[5];
        sum += bfLo(u.w) * w[6] + bfHi(u.w) * w[7];
      }
      sum += __shfl_xor(sum, 1, 4);
      sum += __shfl_xor(sum, 2, 4);
      if (seg == 0)
        Yout[(size_t)s * BATCH + bbase + row] = sum + by;
    }

    // ---- dy = g @ W_X_x^T ; z = dy * (SIGMA * Xn) fused store ----
    v8f acc2[4];
#pragma unroll
    for (int j = 0; j < 4; ++j) acc2[j] = vzero;
    const int mt2 = wave & 3;     // M tile
    const int nh  = wave >> 2;    // N half (0: cols 0-63, 1: cols 64-127)
#pragma unroll 2
    for (int kc = 0; kc < 16; ++kc) {
      const v16bf a = loadA(gS, mt2 * 16, kc * 32, ETA, lane);
#pragma unroll
      for (int j = 0; j < 4; ++j) {
        const v16bf bb = loadB(WXTB, kc * 2 + half, nh * 64 + j * 16 + ncl, 128);
        acc2[j] = wmma_bf16(a, bb, acc2[j]);
      }
    }
#pragma unroll
    for (int j = 0; j < 4; ++j) {
      const int d = nh * 64 + j * 16 + ncl;
      if (d < DD) {
#pragma unroll
        for (int r = 0; r < 8; ++r) {
          const int row = mt2 * 16 + half * 8 + r;
          const float xv = bfLo((unsigned int)xS[row * 128 + d]);
          Zout[(size_t)s * (BATCH * DD) + (size_t)(bbase + row) * DD + d] =
              acc2[j][r] * (SIGMA * xv);
        }
      }
    }
    waitAsync0();       // own wave's async xF writes landed
    __syncthreads();    // everyone's landed; xS/gS reads done -> next step safe
  }
}

// ---------------- launcher ----------------
extern "C" void kernel_launch(void* const* d_in, const int* in_sizes, int n_in,
                              void* d_out, int out_size, void* d_ws,
                              size_t ws_size, hipStream_t stream) {
  (void)in_sizes; (void)n_in; (void)out_size; (void)ws_size;
  const float* t    = (const float*)d_in[0];
  const float* X    = (const float*)d_in[1];
  // d_in[2] = dW (unused by the reference)
  const float* h_m1 = (const float*)d_in[3];
  const float* W_X  = (const float*)d_in[4];
  const float* W_h  = (const float*)d_in[5];
  const float* b_h  = (const float*)d_in[6];
  const float* W_y  = (const float*)d_in[7];
  const float* b_y  = (const float*)d_in[8];

  unsigned int* WhB  = (unsigned int*)d_ws;
  unsigned int* WXB  = WhB + WHB_N;
  unsigned int* WXTB = WXB + WXB_N;

  float* Yout = (float*)d_out;
  float* Zout = Yout + (size_t)NT1 * BATCH;

  const int prepTotal = WHB_N + WXB_N + WXTB_N;   // 196608 = 768 * 256
  prep_kernel<<<prepTotal / TPB, TPB, 0, stream>>>(W_X, W_h, WhB, WXB, WXTB);
  rnn_kernel<<<BATCH / BT, TPB, 0, stream>>>(t, X, h_m1, W_X, b_h, W_y, b_y,
                                             WhB, WXB, WXTB, Yout, Zout);
}
